// GNNDecoder_88201448391207
// MI455X (gfx1250) — compile-verified, matrix operand order checked by reading, not run on hardware
//
#include <hip/hip_runtime.h>
#include <hip/hip_bf16.h>

// ---------------------------------------------------------------------------
// GNN decoder, MI455X (gfx1250) path:
//   pv(f32) -> bf16                                (one pass, halves A traffic)
//   act = softplus(pv @ W1 + b1)    7680x1024x512  bf16 WMMA, f32 accum
//   H   = act @ W2 + b2             7680x512x128   bf16 WMMA, f32 accum
//   P   = H @ Wg[0:128] + pidx term (per-patch, wave reduction)
//   out = 5-point GCN stencil of (P[patch] + pix term) + bg
// GEMMs: 64x128 block tile, K-step 64, double-buffered LDS, async
// global->LDS copies (GLOBAL_LOAD_ASYNC_TO_LDS_B128, ASYNCcnt-tracked).
// ---------------------------------------------------------------------------

typedef __bf16 bf16_t;
typedef __attribute__((ext_vector_type(16))) __bf16 bfx16;
typedef __attribute__((ext_vector_type(8)))  __bf16 bfx8;
typedef __attribute__((ext_vector_type(8)))  float  fx8;
typedef __attribute__((ext_vector_type(4)))  int    i32x4;

union AFrag { bfx16 v; bfx8 h[2]; };

#define M_ROWS 7680
#define K1     1024
#define N1     512
#define K2     512
#define N2     128

#if defined(__HIP_DEVICE_COMPILE__) && \
    __has_builtin(__builtin_amdgcn_global_load_async_to_lds_b128)
#define HAVE_ASYNC_LDS 1
#else
#define HAVE_ASYNC_LDS 0
#endif

#if HAVE_ASYNC_LDS
__device__ __forceinline__ void async_cp16(const void* g, void* l) {
  __builtin_amdgcn_global_load_async_to_lds_b128(
      (__attribute__((address_space(1))) i32x4*)(i32x4*)(const_cast<void*>(g)),
      (__attribute__((address_space(3))) i32x4*)(i32x4*)(l), 0, 0);
}
__device__ __forceinline__ void wait_async() {
#if __has_builtin(__builtin_amdgcn_s_wait_asynccnt)
  __builtin_amdgcn_s_wait_asynccnt(0);
#else
  asm volatile("s_wait_asynccnt 0x0" ::: "memory");
#endif
}
#endif

// ---------------------------------------------------------------- data prep
__global__ __launch_bounds__(256)
void f32_to_bf16_vec(const float* __restrict__ X, bf16_t* __restrict__ Y, int n) {
  int i = (blockIdx.x * 256 + threadIdx.x) * 4;
  if (i >= n) return;
#pragma unroll
  for (int j = 0; j < 4; ++j) Y[i + j] = (bf16_t)X[i + j];
}

__global__ __launch_bounds__(256)
void transpose_to_bf16(const float* __restrict__ W, bf16_t* __restrict__ Wt,
                       int K, int N) {
  // Wt[n][k] = W[k][n]
  int idx = blockIdx.x * 256 + threadIdx.x;
  if (idx >= K * N) return;
  int n = idx / K, k = idx - n * K;
  Wt[idx] = (bf16_t)W[k * N + n];
}

// --------------------------------------------------------------- WMMA GEMM
// out = A[M,K](bf16) @ Bt[N,K]^T(bf16) + bias; SOFTPLUS: softplus -> bf16,
// else f32. Block tile 64(M)x128(N), K-step 64, 8 waves (2Mx4N), wave 32x32.
template <bool SOFTPLUS>
__global__ __launch_bounds__(256)
void gemm_wmma(const bf16_t* __restrict__ A, const bf16_t* __restrict__ Bt,
               const float* __restrict__ bias,
               bf16_t* __restrict__ outBf, float* __restrict__ outF,
               int M, int N, int K)
{
  __shared__ __align__(16) bf16_t Asm[2][64 * 64];    // 16 KB
  __shared__ __align__(16) bf16_t Bsm[2][128 * 64];   // 32 KB

  const int tid  = threadIdx.x;
  const int wave = tid >> 5;
  const int lane = tid & 31;
  const int wm   = wave & 1;        // M sub-tile
  const int wn   = wave >> 1;       // N sub-tile
  const int lrow  = lane & 15;
  const int lhalf = lane >> 4;      // K-half selector per WMMA layout
  const int kA    = lhalf * 8;

  const int m0 = blockIdx.y * 64;
  const int n0 = blockIdx.x * 128;

  fx8 acc[2][2];
#pragma unroll
  for (int mi = 0; mi < 2; ++mi)
#pragma unroll
    for (int ni = 0; ni < 2; ++ni)
#pragma unroll
      for (int r = 0; r < 8; ++r) acc[mi][ni][r] = 0.f;

  // stage one 64-wide K tile (A: 512 x 16B chunks, B: 1024 x 16B chunks)
  auto load_tile = [&](int buf, int k0) {
#if HAVE_ASYNC_LDS
#pragma unroll
    for (int i = 0; i < 2; ++i) {
      int idx = tid + i * 256;              // 0..511
      int r = idx >> 3, c = (idx & 7) * 8;
      async_cp16(&A[(size_t)(m0 + r) * K + k0 + c], &Asm[buf][r * 64 + c]);
    }
#pragma unroll
    for (int i = 0; i < 4; ++i) {
      int idx = tid + i * 256;              // 0..1023
      int r = idx >> 3, c = (idx & 7) * 8;
      async_cp16(&Bt[(size_t)(n0 + r) * K + k0 + c], &Bsm[buf][r * 64 + c]);
    }
#else
#pragma unroll
    for (int i = 0; i < 2; ++i) {
      int idx = tid + i * 256;
      int r = idx >> 3, c = (idx & 7) * 8;
      *(uint4*)&Asm[buf][r * 64 + c] =
          *(const uint4*)&A[(size_t)(m0 + r) * K + k0 + c];
    }
#pragma unroll
    for (int i = 0; i < 4; ++i) {
      int idx = tid + i * 256;
      int r = idx >> 3, c = (idx & 7) * 8;
      *(uint4*)&Bsm[buf][r * 64 + c] =
          *(const uint4*)&Bt[(size_t)(n0 + r) * K + k0 + c];
    }
    if (k0 + 64 < K)   // keep a prefetch hint in the synchronous path
      __builtin_prefetch(&A[(size_t)(m0 + (tid >> 2)) * K + k0 + 64], 0, 1);
#endif
  };

  auto compute = [&](int buf) {
#pragma unroll
    for (int ks = 0; ks < 64; ks += 32) {
      AFrag afr[2], bfr[2];
#pragma unroll
      for (int mi = 0; mi < 2; ++mi) {
        const bf16_t* p = &Asm[buf][(wm * 32 + mi * 16 + lrow) * 64 + ks + kA];
        afr[mi].h[0] = *(const bfx8*)p;
        afr[mi].h[1] = *(const bfx8*)(p + 16);
      }
#pragma unroll
      for (int ni = 0; ni < 2; ++ni) {
        const bf16_t* p = &Bsm[buf][(wn * 32 + ni * 16 + lrow) * 64 + ks + kA];
        bfr[ni].h[0] = *(const bfx8*)p;
        bfr[ni].h[1] = *(const bfx8*)(p + 16);
      }
#pragma unroll
      for (int mi = 0; mi < 2; ++mi)
#pragma unroll
        for (int ni = 0; ni < 2; ++ni)
          acc[mi][ni] = __builtin_amdgcn_wmma_f32_16x16x32_bf16(
              false, afr[mi].v, false, bfr[ni].v, (short)0, acc[mi][ni],
              false, false);
    }
  };

  const int ntiles = K >> 6;
  load_tile(0, 0);
#if HAVE_ASYNC_LDS
  wait_async();
#endif
  __syncthreads();

  for (int t = 0; t < ntiles; ++t) {
    const int cur = t & 1;
    if (t + 1 < ntiles) load_tile(cur ^ 1, (t + 1) << 6);  // overlap with WMMA
    compute(cur);
#if HAVE_ASYNC_LDS
    wait_async();
#endif
    __syncthreads();
  }

#pragma unroll
  for (int mi = 0; mi < 2; ++mi)
#pragma unroll
    for (int ni = 0; ni < 2; ++ni)
#pragma unroll
      for (int r = 0; r < 8; ++r) {
        int grow = m0 + wm * 32 + mi * 16 + (lhalf ? r + 8 : r);
        int gcol = n0 + wn * 32 + ni * 16 + lrow;
        float v  = acc[mi][ni][r] + bias[gcol];
        if (SOFTPLUS) {
          float sp = fmaxf(v, 0.f) + log1pf(expf(-fabsf(v)));
          outBf[(size_t)grow * N + gcol] = (bf16_t)sp;
        } else {
          outF[(size_t)grow * N + gcol] = v;
        }
      }
}

// ------------------------------------------- per-patch projection (wave/row)
__global__ __launch_bounds__(256)
void patch_term_kernel(const float* __restrict__ H, const float* __restrict__ Wg,
                       float* __restrict__ P, int rows)
{
  int gthread = blockIdx.x * blockDim.x + threadIdx.x;
  int row  = gthread >> 5;
  int lane = threadIdx.x & 31;
  if (row >= rows) return;
  float a0 = 0.f, a1 = 0.f, a2 = 0.f;
#pragma unroll
  for (int k = lane; k < 128; k += 32) {
    float hv = H[row * 128 + k];
    a0 += hv * Wg[k * 3 + 0];
    a1 += hv * Wg[k * 3 + 1];
    a2 += hv * Wg[k * 3 + 2];
  }
#pragma unroll
  for (int off = 16; off > 0; off >>= 1) {
    a0 += __shfl_down(a0, off, 32);
    a1 += __shfl_down(a1, off, 32);
    a2 += __shfl_down(a2, off, 32);
  }
  if (lane == 0) {
    int patch = row % 60;
    float px = (float)(patch / 6);
    float py = (float)(patch % 6);
    P[row * 3 + 0] = a0 + px * Wg[128 * 3 + 0] + py * Wg[129 * 3 + 0];
    P[row * 3 + 1] = a1 + px * Wg[128 * 3 + 1] + py * Wg[129 * 3 + 1];
    P[row * 3 + 2] = a2 + px * Wg[128 * 3 + 2] + py * Wg[129 * 3 + 2];
  }
}

// -------------------------------------------------------- GCN 5-point stencil
__device__ __forceinline__ void xw_eval(const float* __restrict__ P,
                                        const float* __restrict__ Wg,
                                        int b, int x, int y, float o[3])
{
  int patch = (x >> 4) * 6 + (y >> 4);
  const float* pt = &P[(b * 60 + patch) * 3];
  float fx = (float)(x & 15) * (1.f / 15.f);
  float fy = (float)(y & 15) * (1.f / 15.f);
#pragma unroll
  for (int c = 0; c < 3; ++c)
    o[c] = pt[c] + fx * Wg[130 * 3 + c] + fy * Wg[131 * 3 + c];
}

__device__ __forceinline__ float deg_of(int x, int y) {
  return 1.f + (float)((x > 0) + (x < 159) + (y > 0) + (y < 95));
}

__global__ __launch_bounds__(256)
void gcn_stencil(const float* __restrict__ P, const float* __restrict__ Wg,
                 const float* __restrict__ bg, float* __restrict__ out)
{
  int x = blockIdx.x * 16 + threadIdx.x;   // 0..159
  int y = blockIdx.y * 16 + threadIdx.y;   // 0..95
  int b = blockIdx.z;                      // 0..127

  float degs  = deg_of(x, y);
  float dself = rsqrtf(degs);
  float invd  = 1.f / degs;

  float xs[3];
  xw_eval(P, Wg, b, x, y, xs);
  float acc[3] = {xs[0] * invd, xs[1] * invd, xs[2] * invd};

  const int dx[4] = {-1, 1, 0, 0};
  const int dy[4] = {0, 0, -1, 1};
#pragma unroll
  for (int d = 0; d < 4; ++d) {
    int nx = x + dx[d], ny = y + dy[d];
    if (nx < 0 || nx > 159 || ny < 0 || ny > 95) continue;
    float xn[3];
    xw_eval(P, Wg, b, nx, ny, xn);
    float w = rsqrtf(deg_of(nx, ny)) * dself;
#pragma unroll
    for (int c = 0; c < 3; ++c) acc[c] += xn[c] * w;
  }

  long base = (((long)b * 160 + x) * 96 + y) * 3;
#pragma unroll
  for (int c = 0; c < 3; ++c) out[base + c] = acc[c] + bg[c];
}

// ---------------------------------------------------------------------------
extern "C" void kernel_launch(void* const* d_in, const int* in_sizes, int n_in,
                              void* d_out, int out_size, void* d_ws, size_t ws_size,
                              hipStream_t stream)
{
  const float* pv = (const float*)d_in[0];   // (7680, 1024)
  const float* W1 = (const float*)d_in[1];   // (1024, 512)
  const float* b1 = (const float*)d_in[2];   // (512,)
  const float* W2 = (const float*)d_in[3];   // (512, 128)
  const float* b2 = (const float*)d_in[4];   // (128,)
  const float* Wg = (const float*)d_in[5];   // (132, 3)
  const float* bg = (const float*)d_in[6];   // (3,)
  // d_in[7] = edge_index: 4-neighbor grid is known analytically, unused.

  char* ws = (char*)d_ws;
  size_t off = 0;
  bf16_t* pvB = (bf16_t*)(ws + off); off += (size_t)M_ROWS * K1 * sizeof(bf16_t); // 15.7 MB
  bf16_t* W1t = (bf16_t*)(ws + off); off += (size_t)N1 * K1 * sizeof(bf16_t);     // 1 MB
  bf16_t* W2t = (bf16_t*)(ws + off); off += (size_t)N2 * K2 * sizeof(bf16_t);     // 128 KB
  bf16_t* act = (bf16_t*)(ws + off); off += (size_t)M_ROWS * N1 * sizeof(bf16_t); // 7.9 MB
  float*  H   = (float*) (ws + off); off += (size_t)M_ROWS * N2 * sizeof(float);  // 3.9 MB
  float*  P   = (float*) (ws + off); off += (size_t)M_ROWS * 3  * sizeof(float);  // 92 KB

  f32_to_bf16_vec<<<(M_ROWS * K1 / 4 + 255) / 256, 256, 0, stream>>>(
      pv, pvB, M_ROWS * K1);
  transpose_to_bf16<<<(K1 * N1 + 255) / 256, 256, 0, stream>>>(W1, W1t, K1, N1);
  transpose_to_bf16<<<(K2 * N2 + 255) / 256, 256, 0, stream>>>(W2, W2t, K2, N2);

  dim3 g1(N1 / 128, M_ROWS / 64);   // (4, 120)
  gemm_wmma<true><<<g1, 256, 0, stream>>>(pvB, W1t, b1, act, (float*)nullptr,
                                          M_ROWS, N1, K1);

  dim3 g2(N2 / 128, M_ROWS / 64);   // (1, 120)
  gemm_wmma<false><<<g2, 256, 0, stream>>>(act, W2t, b2, (bf16_t*)nullptr, H,
                                           M_ROWS, N2, K2);

  patch_term_kernel<<<(M_ROWS * 32 + 255) / 256, 256, 0, stream>>>(H, Wg, P,
                                                                   M_ROWS);

  dim3 gs(10, 6, 128);
  gcn_stencil<<<gs, dim3(16, 16, 1), 0, stream>>>(P, Wg, bg, (float*)d_out);
}
